// MACAttention_90829968376400
// MI455X (gfx1250) — matrix operand order repaired and checked by reading, not compiled, not run on hardware
//
#include <hip/hip_runtime.h>
#include <hip/hip_bf16.h>

typedef _Float16 h16;
typedef __attribute__((ext_vector_type(16))) _Float16 v16h;
typedef __attribute__((ext_vector_type(8)))  _Float16 v8h;
typedef __attribute__((ext_vector_type(8)))  float    v8f;
typedef __attribute__((ext_vector_type(4)))  unsigned u32x4;
typedef __attribute__((ext_vector_type(8)))  unsigned u32x8;

#define HEADS   16
#define DHEAD   128
#define DIM     2048
#define SEG     512
#define KPAD    544            // 16 persist + 512 keys + 16 pad
#define BATCH   2
#define SEQ     4096
#define ROWS    (BATCH*SEQ)    // 8192
#define WINS    16             // BATCH * (SEQ/SEG)
#define NQKV    (3*DIM)        // 6144
#define NEGBIG  (-1.0e30f)

__device__ __forceinline__ v8f v8f_zero() {
  v8f z;
#pragma unroll
  for (int i = 0; i < 8; i++) z[i] = 0.f;
  return z;
}

__device__ __forceinline__ v16h make_v16h(v8h lo, v8h up) {
  v16h r;
#pragma unroll
  for (int i = 0; i < 8; i++) { r[i] = lo[i]; r[i + 8] = up[i]; }
  return r;
}

__device__ __forceinline__ v8f wmma_fp16(v16h a, v16h b, v8f c) {
  // D = A(16x32 f16) * B(32x16 f16) + C(16x16 f32)
  return __builtin_amdgcn_wmma_f32_16x16x32_f16(
      /*neg_a=*/false, a, /*neg_b=*/false, b,
      /*c_mod=*/(short)0, c, /*reuse_a=*/false, /*reuse_b=*/false);
}

// LDS byte offset of a __shared__ object: LDS aperture maps addr[31:0] -> LDS offset.
__device__ __forceinline__ unsigned lds_off(const void* p) {
  return (unsigned)(unsigned long long)p;
}

// Async global->LDS copy of 32 bytes per lane (2 x b128), no VGPR data path.
__device__ __forceinline__ void async_copy32(unsigned ldsAddr, const h16* gptr) {
  asm volatile("global_load_async_to_lds_b128 %0, %1, off"
               :: "v"(ldsAddr), "v"(gptr) : "memory");
  asm volatile("global_load_async_to_lds_b128 %0, %1, off offset:16"
               :: "v"(ldsAddr), "v"(gptr) : "memory");
}

__device__ __forceinline__ void wait_asynccnt0() { asm volatile("s_wait_asynccnt 0x0" ::: "memory"); }
__device__ __forceinline__ void wait_asynccnt2() { asm volatile("s_wait_asynccnt 0x2" ::: "memory"); }
__device__ __forceinline__ void wait_asynccnt4() { asm volatile("s_wait_asynccnt 0x4" ::: "memory"); }

// Tensor Data Mover: 1-D contiguous DMA of `elems` f16 from global to LDS.
__device__ __forceinline__ void tdm_load_1d(unsigned ldsAddr, const h16* gptr, unsigned elems) {
  unsigned long long ga = (unsigned long long)gptr;
  u32x4 g0;
  g0[0] = 1u;                                            // count = 1 valid descriptor
  g0[1] = ldsAddr;                                       // lds_addr (bytes)
  g0[2] = (unsigned)(ga & 0xFFFFFFFFu);                  // global_addr[31:0]
  g0[3] = (unsigned)((ga >> 32) & 0x1FFFFFFu)            // global_addr[56:32]
        | (2u << 30);                                    // type = 2 ("image")
  u32x8 g1;
  g1[0] = 0x00010000u;                                   // workgroup_mask=0, data_size=1 (2B)
  g1[1] = (elems & 0xFFFFu) << 16;                       // tensor_dim0[15:0] @ bits 79:48
  g1[2] = (elems >> 16) | (1u << 16);                    // tensor_dim0[31:16] | tensor_dim1=1
  g1[3] = (elems & 0xFFFFu) << 16;                       // tile_dim0 @ bits 127:112
  g1[4] = 1u;                                            // tile_dim1 = 1
  g1[5] = elems;                                         // tensor_dim0_stride[31:0]
  g1[6] = 0u;
  g1[7] = 0u;
  asm volatile("tensor_load_to_lds %0, %1" :: "s"(g0), "s"(g1) : "memory");
}

// ---------------------------------------------------------------------------
// 1) RMSNorm: fp32 row -> f16 normalized row
// ---------------------------------------------------------------------------
__global__ __launch_bounds__(256) void rmsnorm_kernel(
    const float* __restrict__ x, const float* __restrict__ w, h16* __restrict__ xn)
{
  const int row = blockIdx.x;
  const float* xr = x + (size_t)row * DIM;
  float ss = 0.f;
  for (int i = threadIdx.x; i < DIM; i += 256) { float v = xr[i]; ss += v * v; }
#pragma unroll
  for (int o = 16; o > 0; o >>= 1) ss += __shfl_xor(ss, o, 32);
  __shared__ float red[8];
  __shared__ float stot;
  if ((threadIdx.x & 31) == 0) red[threadIdx.x >> 5] = ss;
  __syncthreads();
  if (threadIdx.x == 0) {
    float t = 0.f;
    for (int i = 0; i < 8; i++) t += red[i];
    stot = t;
  }
  __syncthreads();
  const float rs = rsqrtf(stot / (float)DIM + 1e-6f);
  for (int i = threadIdx.x; i < DIM; i += 256)
    xn[(size_t)row * DIM + i] = (h16)(xr[i] * rs * w[i]);
}

// ---------------------------------------------------------------------------
// 2) fp32 -> f16 conversion (weights)
// ---------------------------------------------------------------------------
__global__ void f32_to_f16_kernel(const float* __restrict__ in, h16* __restrict__ out, int n)
{
  int i = blockIdx.x * 256 + threadIdx.x;
  if (i < n) out[i] = (h16)in[i];
}

// ---------------------------------------------------------------------------
// 3/7) WMMA GEMM: D[M,N] = A[M,K](f16) * B[N,K]^T(f16), f32 accumulate.
//     Block = 128 threads (4 waves), tile 64x64, K-step 32.
//     Double-buffered async global->LDS staging: tile i+1's DMAs stay in
//     flight (ASYNCcnt<=4) while tile i runs on the WMMA pipe.
// ---------------------------------------------------------------------------
template <typename OutT>
__global__ __launch_bounds__(128) void gemm_wmma_kernel(
    const h16* __restrict__ A, const h16* __restrict__ Bw,
    OutT* __restrict__ D, int Ncols, int K)
{
  __shared__ __attribute__((aligned(32))) h16 lA[2][64 * 32];
  __shared__ __attribute__((aligned(32))) h16 lB[2][64 * 32];

  const int tid  = threadIdx.x;
  const int wave = tid >> 5;
  const int lane = tid & 31;
  const int l16  = lane & 15;
  const int hi   = lane >> 4;

  const int mBase = blockIdx.y * 64;
  const int nBase = blockIdx.x * 64;

  const int ldr = tid >> 1;   // 0..63 : tile row being staged
  const int seg = tid & 1;    // half of the 32-wide K slice

  const unsigned tOff = (unsigned)((ldr * 32 + seg * 16) * sizeof(h16));
  const unsigned lAoff[2] = { lds_off(&lA[0][0]) + tOff, lds_off(&lA[1][0]) + tOff };
  const unsigned lBoff[2] = { lds_off(&lB[0][0]) + tOff, lds_off(&lB[1][0]) + tOff };

  const h16* gA = A  + (size_t)(mBase + ldr) * K + seg * 16;
  const h16* gB = Bw + (size_t)(nBase + ldr) * K + seg * 16;

  v8f acc[4];
#pragma unroll
  for (int j = 0; j < 4; j++) acc[j] = v8f_zero();

  // prologue: stage tile 0 into buffer 0
  async_copy32(lAoff[0], gA);
  async_copy32(lBoff[0], gB);

  int buf = 0;
  for (int k0 = 0; k0 < K; k0 += 32) {
    const bool hasNext = (k0 + 32) < K;
    if (hasNext) {                       // prefetch tile i+1 into the other buffer
      async_copy32(lAoff[buf ^ 1], gA + k0 + 32);
      async_copy32(lBoff[buf ^ 1], gB + k0 + 32);
      wait_asynccnt4();                  // drain tile i, leave tile i+1 in flight
    } else {
      wait_asynccnt0();
    }
    __syncthreads();

    // A fragment: lane<16 -> row=lane, K 0..15; lane>=16 -> K 16..31 (contiguous 32B)
    const h16* arow = &lA[buf][(wave * 16 + l16) * 32];
    v16h af = *reinterpret_cast<const v16h*>(arow + hi * 16);
#pragma unroll
    for (int j = 0; j < 4; j++) {
      // B fragment: lane<16 -> col=lane, K 0..15; lane>=16 -> K 16..31 (contiguous 32B)
      v16h bf = *reinterpret_cast<const v16h*>(&lB[buf][(j * 16 + l16) * 32 + hi * 16]);
      acc[j] = wmma_fp16(af, bf, acc[j]);
    }
    __syncthreads();                     // retire readers before buffer is re-filled
    buf ^= 1;
  }

  // C/D layout: lane holds col l16; VGPR r holds row r (+8 for upper half-wave)
#pragma unroll
  for (int j = 0; j < 4; j++) {
    const int col = nBase + j * 16 + l16;
#pragma unroll
    for (int r = 0; r < 8; r++) {
      const int row = mBase + wave * 16 + r + hi * 8;
      D[(size_t)row * Ncols + col] = (OutT)acc[j][r];
    }
  }
}

// ---------------------------------------------------------------------------
// 4) RoPE + scatter into per-(window,head) attention layouts.
//    q -> [wi][h][512][128]; k -> [wi][h][544][128] at row 16+qr;
//    v -> transposed [wi][h][128][544] (dim-major) for contiguous PV B-frags.
// ---------------------------------------------------------------------------
__global__ __launch_bounds__(256) void rope_scatter_kernel(
    const h16* __restrict__ qkv, h16* __restrict__ qw,
    h16* __restrict__ kw, h16* __restrict__ vT)
{
  const int row = blockIdx.x;          // 0..8191
  const int pos = row & (SEQ - 1);
  const int b_i = row >> 12;
  const int wi  = b_i * 8 + (pos >> 9);
  const int qr  = pos & (SEG - 1);

  const h16* src = qkv + (size_t)row * NQKV;
  const int c0 = threadIdx.x * 8;      // 8 elements (4 rope pairs) per thread
  const int h  = c0 >> 7;
  const int d0 = c0 & 127;

  float vq[8], vk[8], vv[8];
#pragma unroll
  for (int i = 0; i < 8; i++) {
    vq[i] = (float)src[c0 + i];
    vk[i] = (float)src[2048 + c0 + i];
    vv[i] = (float)src[4096 + c0 + i];
  }
  const float LOG_TH = 9.210340371976184f;   // ln(10000)
#pragma unroll
  for (int p = 0; p < 4; p++) {
    const int de = d0 + 2 * p;
    const float inv = __expf(-(float)de * (LOG_TH / 128.0f));
    const float ang = (float)pos * inv;
    float s, c;
    __sincosf(ang, &s, &c);
    float q0 = vq[2*p], q1 = vq[2*p+1];
    vq[2*p]   = q0 * c - q1 * s;
    vq[2*p+1] = q1 * c + q0 * s;
    float k0 = vk[2*p], k1 = vk[2*p+1];
    vk[2*p]   = k0 * c - k1 * s;
    vk[2*p+1] = k1 * c + k0 * s;
  }

  h16* qdst = qw + (((size_t)wi * HEADS + h) * SEG  + qr)      * DHEAD + d0;
  h16* kdst = kw + (((size_t)wi * HEADS + h) * KPAD + 16 + qr) * DHEAD + d0;
#pragma unroll
  for (int i = 0; i < 8; i++) { qdst[i] = (h16)vq[i]; kdst[i] = (h16)vk[i]; }
  h16* vbase = vT + (((size_t)wi * HEADS + h) * DHEAD) * KPAD;
#pragma unroll
  for (int i = 0; i < 8; i++) vbase[(size_t)(d0 + i) * KPAD + (16 + qr)] = (h16)vv[i];
}

// ---------------------------------------------------------------------------
// 5) Persistent-memory prefix (rows 0..15) + zero pad (rows 528..543)
// ---------------------------------------------------------------------------
__global__ __launch_bounds__(128) void persist_pad_kernel(
    const float* __restrict__ pm, h16* __restrict__ kw, h16* __restrict__ vT)
{
  const int wh = blockIdx.x;          // wi*16 + h
  const int h  = wh & 15;
  const int d  = threadIdx.x;         // 0..127
  h16* kbase = kw + (size_t)wh * KPAD * DHEAD;
  h16* vbase = vT + (size_t)wh * DHEAD * KPAD;
#pragma unroll
  for (int kr = 0; kr < 16; kr++) {
    const float kval = pm[(((0 * HEADS + h) * 16) + kr) * DHEAD + d];
    const float vval = pm[(((1 * HEADS + h) * 16) + kr) * DHEAD + d];
    kbase[kr * DHEAD + d] = (h16)kval;
    vbase[(size_t)d * KPAD + kr] = (h16)vval;
  }
#pragma unroll
  for (int kr = 528; kr < KPAD; kr++) {
    kbase[kr * DHEAD + d] = (h16)0.f;
    vbase[(size_t)d * KPAD + kr] = (h16)0.f;
  }
}

// ---------------------------------------------------------------------------
// 6) Flash attention per (window, head): 256 thr = 8 waves x 16 q-rows.
//    17 key blocks of 32, double-buffered: K block via Tensor Data Mover
//    (TENSORcnt<=1), V block via async global->LDS (ASYNCcnt<=2); block i+1's
//    DMAs overlap block i's WMMA/softmax work.
// ---------------------------------------------------------------------------
__global__ __launch_bounds__(256) void attn_kernel(
    const h16* __restrict__ qw, const h16* __restrict__ kw,
    const h16* __restrict__ vT, h16* __restrict__ aout)
{
  __shared__ __attribute__((aligned(32))) h16 lk[2][32 * 128];  // [key][dim]
  __shared__ __attribute__((aligned(32))) h16 lv[2][128 * 32];  // [dim][key]
  __shared__ __attribute__((aligned(32))) h16 lp[8][16 * 32];   // per-wave P scratch

  const int tid = threadIdx.x, wave = tid >> 5, lane = tid & 31;
  const int l16 = lane & 15, hi = lane >> 4;
  const int qc = blockIdx.x, h = blockIdx.y, wi = blockIdx.z;
  const int qbase = qc * 128 + wave * 16;
  const size_t wh = (size_t)wi * HEADS + h;
  const int ko = hi ? 8 : 0;

  // Q fragments: 4 K-chunks of 32
  const h16* qp = qw + (wh * SEG + qbase) * DHEAD;
  v16h qf[4];
#pragma unroll
  for (int kc = 0; kc < 4; kc++) {
    const h16* qrow = qp + l16 * DHEAD + kc * 32;
    qf[kc] = *reinterpret_cast<const v16h*>(qrow + hi * 16);
  }

  float mrow[8], lrow[8];
  v8f oacc[8];
#pragma unroll
  for (int r = 0; r < 8; r++) { mrow[r] = NEGBIG; lrow[r] = 0.f; }
#pragma unroll
  for (int j = 0; j < 8; j++) oacc[j] = v8f_zero();

  const h16* kbaseg = kw + wh * KPAD * DHEAD;
  const h16* vbaseg = vT + wh * DHEAD * KPAD;

  const unsigned lkBase[2] = { lds_off(&lk[0][0]), lds_off(&lk[1][0]) };
  const int dd = tid >> 1;    // 0..127 dims (V staging)
  const int s2 = tid & 1;
  const unsigned vOff = (unsigned)((dd * 32 + s2 * 16) * sizeof(h16));
  const unsigned lvOff[2] = { lds_off(&lv[0][0]) + vOff, lds_off(&lv[1][0]) + vOff };
  const h16* gV = vbaseg + (size_t)dd * KPAD + s2 * 16;

  // prologue: stage key block 0 into buffer 0
  async_copy32(lvOff[0], gV);
  if (wave == 0) tdm_load_1d(lkBase[0], kbaseg, 32 * DHEAD);

  int buf = 0;
  for (int kb = 0; kb < 17; kb++) {
    const bool hasNext = kb < 16;
    if (hasNext) {                       // prefetch block kb+1 into the other buffer
      async_copy32(lvOff[buf ^ 1], gV + (kb + 1) * 32);
      if (wave == 0)
        tdm_load_1d(lkBase[buf ^ 1], kbaseg + (size_t)(kb + 1) * 32 * DHEAD, 32 * DHEAD);
      wait_asynccnt2();                  // drain block kb, leave kb+1 in flight
      __builtin_amdgcn_s_wait_tensorcnt(1);
    } else {
      wait_asynccnt0();
      __builtin_amdgcn_s_wait_tensorcnt(0);
    }
    __syncthreads();

    // S = Q * K^T  (two 16x16 tiles covering 32 keys)
    v8f s0 = v8f_zero(), s1 = v8f_zero();
#pragma unroll
    for (int kc = 0; kc < 4; kc++) {
      v16h b0 = *reinterpret_cast<const v16h*>(&lk[buf][(l16)      * 128 + kc * 32 + hi * 16]);
      v16h b1 = *reinterpret_cast<const v16h*>(&lk[buf][(16 + l16) * 128 + kc * 32 + hi * 16]);
      s0 = wmma_fp16(qf[kc], b0, s0);
      s1 = wmma_fp16(qf[kc], b1, s1);
    }

    // scale + MAC attention bias (prefix 32 free, causal on segment, pad masked)
    const float scale = 0.08838834764831845f;   // 1/sqrt(128)
    const int n0 = kb * 32 + l16;
    const int n1 = n0 + 16;
    float bm[8], psum[8], alpha[8];
#pragma unroll
    for (int r = 0; r < 8; r++) {
      const int qi = qbase + r + hi * 8;
      const bool a0 = (n0 < 32) || ((n0 < 528) && (qi >= 16) && ((n0 - 32) <= (qi - 16)));
      const bool a1 = (n1 < 32) || ((n1 < 528) && (qi >= 16) && ((n1 - 32) <= (qi - 16)));
      s0[r] = s0[r] * scale + (a0 ? 0.f : NEGBIG);
      s1[r] = s1[r] * scale + (a1 ? 0.f : NEGBIG);
      bm[r] = fmaxf(s0[r], s1[r]);
    }
#pragma unroll
    for (int o = 1; o < 16; o <<= 1)
#pragma unroll
      for (int r = 0; r < 8; r++) bm[r] = fmaxf(bm[r], __shfl_xor(bm[r], o, 32));
#pragma unroll
    for (int r = 0; r < 8; r++) {
      const float nm = fmaxf(mrow[r], bm[r]);
      alpha[r] = __expf(mrow[r] - nm);
      mrow[r]  = nm;
      s0[r] = __expf(s0[r] - nm);
      s1[r] = __expf(s1[r] - nm);
      psum[r] = s0[r] + s1[r];
    }
#pragma unroll
    for (int o = 1; o < 16; o <<= 1)
#pragma unroll
      for (int r = 0; r < 8; r++) psum[r] += __shfl_xor(psum[r], o, 32);
#pragma unroll
    for (int r = 0; r < 8; r++) lrow[r] = lrow[r] * alpha[r] + psum[r];
#pragma unroll
    for (int j = 0; j < 8; j++)
#pragma unroll
      for (int r = 0; r < 8; r++) oacc[j][r] *= alpha[r];

    // P: C/D layout fp32 -> per-wave LDS -> A-layout f16 fragment
    h16* ps = &lp[wave][0];
#pragma unroll
    for (int r = 0; r < 8; r++) {
      const int m = r + hi * 8;
      ps[m * 32 + l16]      = (h16)s0[r];
      ps[m * 32 + 16 + l16] = (h16)s1[r];
    }
    const h16* prow = ps + l16 * 32;
    v16h pa = *reinterpret_cast<const v16h*>(prow + hi * 16);

    // O += P[16x32] * V[32x128]
#pragma unroll
    for (int j = 0; j < 8; j++) {
      v16h bv = *reinterpret_cast<const v16h*>(&lv[buf][(j * 16 + l16) * 32 + hi * 16]);
      oacc[j] = wmma_fp16(pa, bv, oacc[j]);
    }
    __syncthreads();                     // retire readers before buffer is re-filled
    buf ^= 1;
  }

  // normalize and store to [b, seqpos, h*128] f16
  const int b_i = wi >> 3, w_i = wi & 7;
#pragma unroll
  for (int r = 0; r < 8; r++) {
    const int qi = qbase + r + hi * 8;
    const size_t row = (size_t)b_i * SEQ + (size_t)w_i * SEG + qi;
    const float inv = 1.0f / lrow[r];
#pragma unroll
    for (int j = 0; j < 8; j++)
      aout[row * DIM + h * DHEAD + j * 16 + l16] = (h16)(oacc[j][r] * inv);
  }
}

// ---------------------------------------------------------------------------
extern "C" void kernel_launch(void* const* d_in, const int* in_sizes, int n_in,
                              void* d_out, int out_size, void* d_ws, size_t ws_size,
                              hipStream_t stream)
{
  (void)in_sizes; (void)n_in; (void)out_size; (void)ws_size;
  const float* x      = (const float*)d_in[0];
  const float* norm_w = (const float*)d_in[1];
  const float* w_qkv  = (const float*)d_in[2];
  const float* w_out  = (const float*)d_in[3];
  const float* pm     = (const float*)d_in[4];
  float* out = (float*)d_out;

  h16* ws    = (h16*)d_ws;
  h16* xn    = ws;
  h16* wqkvh = xn    + (size_t)ROWS * DIM;
  h16* wouth = wqkvh + (size_t)NQKV * DIM;
  h16* qkv   = wouth + (size_t)DIM * DIM;
  h16* qwb   = qkv   + (size_t)ROWS * NQKV;
  h16* kwb   = qwb   + (size_t)WINS * HEADS * SEG * DHEAD;
  h16* vTb   = kwb   + (size_t)WINS * HEADS * KPAD * DHEAD;
  h16* aoutb = vTb   + (size_t)WINS * HEADS * DHEAD * KPAD;

  rmsnorm_kernel<<<ROWS, 256, 0, stream>>>(x, norm_w, xn);
  f32_to_f16_kernel<<<(NQKV * DIM + 255) / 256, 256, 0, stream>>>(w_qkv, wqkvh, NQKV * DIM);
  f32_to_f16_kernel<<<(DIM * DIM + 255) / 256, 256, 0, stream>>>(w_out, wouth, DIM * DIM);
  gemm_wmma_kernel<h16><<<dim3(NQKV / 64, ROWS / 64), 128, 0, stream>>>(xn, wqkvh, qkv, NQKV, DIM);
  rope_scatter_kernel<<<ROWS, 256, 0, stream>>>(qkv, qwb, kwb, vTb);
  persist_pad_kernel<<<WINS * HEADS, 128, 0, stream>>>(pm, kwb, vTb);
  attn_kernel<<<dim3(SEG / 128, HEADS, WINS), 256, 0, stream>>>(qwb, kwb, vTb, aoutb);
  gemm_wmma_kernel<float><<<dim3(DIM / 64, ROWS / 64), 128, 0, stream>>>(aoutb, wouth, out, DIM, DIM);
}
